// Gate_25537875542561
// MI455X (gfx1250) — compile-verified, matrix operand order checked by reading, not run on hardware
//
#include <hip/hip_runtime.h>
#include <hip/hip_bf16.h>
#include <math.h>

// ---------------------------------------------------------------------------
// DeepSeek-style router gate for MI455X (gfx1250, wave32, WMMA, async-LDS).
//
// Roofline: router GEMM 16384x7168 @ 7168x256 = 60.2 GFLOP; x = 470 MB
// (HBM floor ~20us @ 23.3 TB/s) -> x must be read exactly ONCE (BN=256).
// Weight = 7.3 MB -> L2-resident; pre-split once to bf16 hi/lo so GEMM can
// stage B tiles via gfx1250 async global->LDS copies (no VALU), while the
// loader lanes split x fp32->bf16 hi/lo. 3-term bf16 split WMMA gives
// fp32-faithful logits (needed: top-k ranking is precision sensitive).
// ---------------------------------------------------------------------------

#define TOKENS 16384
#define DIM    7168
#define NEXP   256
#define NGROUP 8
#define TOPKG  4
#define TOPK   8
#define RSCALE 2.5f

typedef __attribute__((ext_vector_type(16))) __bf16 v16bf;
typedef __attribute__((ext_vector_type(8)))  float  v8f;

#define BM  64      // tokens per workgroup
#define BN  256     // all experts -> x streamed from HBM exactly once
#define BK  32      // K chunk (one WMMA K step)
#define LDK 40      // padded LDS row stride (elems): 80B rows -> 16B-aligned
                    // fragments, destaggered banks across a 16-row fragment

// ---------------------------------------------------------------------------
// Prep: split weight fp32 -> bf16 hi/lo once (7.3 MB, ~1us). Removes all
// weight-conversion VALU from the GEMM hot loop.
// ---------------------------------------------------------------------------
__global__ __launch_bounds__(256) void weight_split_kernel(
    const float* __restrict__ w, __bf16* __restrict__ w_hi,
    __bf16* __restrict__ w_lo)
{
  size_t i4 = (size_t)blockIdx.x * blockDim.x + threadIdx.x;   // float4 index
  float4 f = *(const float4*)(w + i4 * 4);
  float fs[4] = {f.x, f.y, f.z, f.w};
#pragma unroll
  for (int i = 0; i < 4; ++i) {
    __bf16 h = (__bf16)fs[i];
    w_hi[i4 * 4 + i] = h;
    w_lo[i4 * 4 + i] = (__bf16)(fs[i] - (float)h);
  }
}

__device__ __forceinline__ void split4(__bf16* hi, __bf16* lo, int base, float4 f) {
  float fs[4] = {f.x, f.y, f.z, f.w};
#pragma unroll
  for (int i = 0; i < 4; ++i) {
    float v = fs[i];
    __bf16 h = (__bf16)v;
    hi[base + i] = h;
    lo[base + i] = (__bf16)(v - (float)h);
  }
}

// lds_off: byte address in LDS space (low 32 bits of a generic LDS pointer,
// per ISA aperture rule addr[31:0] == ds offset). 16B per lane per op,
// tracked with ASYNCcnt.
__device__ __forceinline__ void async_copy_b128(uint32_t lds_off, uint64_t gaddr) {
  asm volatile("global_load_async_to_lds_b128 %0, %1, off"
               :: "v"(lds_off), "v"(gaddr) : "memory");
}

__global__ __launch_bounds__(256) void router_gemm_kernel(
    const float* __restrict__ x,
    const __bf16* __restrict__ w_hi, const __bf16* __restrict__ w_lo,
    float* __restrict__ logits)
{
  __shared__ __align__(16) __bf16 a_hi[BM * LDK];
  __shared__ __align__(16) __bf16 a_lo[BM * LDK];
  __shared__ __align__(16) __bf16 b_hi[BN * LDK];
  __shared__ __align__(16) __bf16 b_lo[BN * LDK];

  const int tid  = threadIdx.x;
  const int lane = tid & 31;
  const int wave = tid >> 5;
  const int m0   = blockIdx.x * BM;

  const int strip = wave >> 1;          // 0..3 : 16-row M strip
  const int nhalf = wave & 1;           // 0..1 : 128-expert half
  const int mrow  = strip * 16 + (lane & 15);
  const int koff  = (lane < 16) ? 0 : 16;   // lanes 16-31 hold K+16

  const uint32_t lds_bhi = (uint32_t)(uintptr_t)(&b_hi[0]);
  const uint32_t lds_blo = (uint32_t)(uintptr_t)(&b_lo[0]);

  v8f acc[8] = {};                      // 8 tiles of 16x16 f32 (64 VGPRs)

  for (int k0 = 0; k0 < DIM; k0 += BK) {
    // --- B tiles: async copy prepacked bf16 hi/lo, 256 rows x 64B each.
    //     1024 x 16B chunks per buffer; 4 chunks/thread/buffer.
#pragma unroll
    for (int v = 0; v < 4; ++v) {
      int c   = tid + v * 256;          // chunk id 0..1023
      int row = c >> 2;                 // expert row
      int q   = c & 3;                  // 16B quarter within the 64B row
      uint32_t loff  = (uint32_t)(row * (LDK * 2) + q * 16);
      uint64_t gbyte = ((uint64_t)row * DIM + (uint64_t)k0) * 2 + (uint64_t)q * 16;
      async_copy_b128(lds_bhi + loff, (uint64_t)(uintptr_t)w_hi + gbyte);
      async_copy_b128(lds_blo + loff, (uint64_t)(uintptr_t)w_lo + gbyte);
    }
    // --- A tile: 64x32 f32, split to bf16 hi/lo on the VALU (overlaps the
    //     async B copies). 512 float4 -> 2/thread.
#pragma unroll
    for (int v = 0; v < 2; ++v) {
      int j   = tid + v * 256;
      int row = j >> 3;
      int c4  = (j & 7) * 4;
      float4 f = *(const float4*)(x + (size_t)(m0 + row) * DIM + k0 + c4);
      split4(a_hi, a_lo, row * LDK + c4, f);
    }
    asm volatile("s_wait_asynccnt 0" ::: "memory");
    __syncthreads();

    // A fragment: per-lane 16 contiguous K of row `mrow`
    v16bf ah = *(const v16bf*)(a_hi + mrow * LDK + koff);
    v16bf al = *(const v16bf*)(a_lo + mrow * LDK + koff);

#pragma unroll
    for (int nt = 0; nt < 8; ++nt) {
      int brow = nhalf * 128 + nt * 16 + (lane & 15);    // expert (= B col)
      v16bf bh = *(const v16bf*)(b_hi + brow * LDK + koff);
      v16bf bl = *(const v16bf*)(b_lo + brow * LDK + koff);
      // fp32 ~ hi*hi + lo*hi + hi*lo  (lo*lo ~2^-16, dropped)
      acc[nt] = __builtin_amdgcn_wmma_f32_16x16x32_bf16(
          false, ah, false, bh, (short)0, acc[nt], false, false);
      acc[nt] = __builtin_amdgcn_wmma_f32_16x16x32_bf16(
          false, al, false, bh, (short)0, acc[nt], false, false);
      acc[nt] = __builtin_amdgcn_wmma_f32_16x16x32_bf16(
          false, ah, false, bl, (short)0, acc[nt], false, false);
    }
    __syncthreads();
  }

  // C/D layout: lanes 0-15 N=lane, M=r ; lanes 16-31 N=lane-16, M=8+r
  const int mbase = m0 + strip * 16 + ((lane >> 4) << 3);
#pragma unroll
  for (int nt = 0; nt < 8; ++nt) {
    int N = nhalf * 128 + nt * 16 + (lane & 15);
#pragma unroll
    for (int r = 0; r < 8; ++r) {
      logits[(size_t)(mbase + r) * NEXP + N] = acc[nt][r];
    }
  }
}

// ---------------------------------------------------------------------------
// Gate: one wave32 per token. Lane L owns experts [8L, 8L+8) -> all inside
// group L>>2 (4 lanes per group of 32 experts).
// ---------------------------------------------------------------------------
__global__ __launch_bounds__(256) void gate_topk_kernel(
    const float* __restrict__ logits, const float* __restrict__ bias,
    float* __restrict__ out_w, int* __restrict__ out_idx)
{
  const int lane  = threadIdx.x & 31;
  const int wave  = threadIdx.x >> 5;
  const int token = blockIdx.x * 8 + wave;

  float s[8], b[8];
#pragma unroll
  for (int j = 0; j < 8; ++j) {
    int e   = lane * 8 + j;
    float z = logits[(size_t)token * NEXP + e];
    float sg = 1.0f / (1.0f + __expf(-z));     // sigmoid
    s[j] = sg;                                  // original score (for weights)
    b[j] = sg + bias[e];                        // biased score (for routing)
  }

  // per-lane top-2 of biased scores
  float t1 = -INFINITY, t2 = -INFINITY;
#pragma unroll
  for (int j = 0; j < 8; ++j) {
    float v = b[j];
    if (v > t1) { t2 = t1; t1 = v; }
    else if (v > t2) { t2 = v; }
  }
  // merge top-2 across the 4 lanes of each group (quad butterfly)
#pragma unroll
  for (int m = 1; m <= 2; m <<= 1) {
    float o1 = __shfl_xor(t1, m, 32);
    float o2 = __shfl_xor(t2, m, 32);
    float n1 = fmaxf(t1, o1);
    float n2 = fmaxf(fminf(t1, o1), fmaxf(t2, o2));
    t1 = n1; t2 = n2;
  }
  float gs = t1 + t2;                           // group score (uniform in quad)

  // gather all 8 group scores, rank-select top-4 groups (low index wins ties)
  float g[8];
#pragma unroll
  for (int gg = 0; gg < 8; ++gg) g[gg] = __shfl(gs, gg * 4, 32);
  int mg = lane >> 2;
  int rank = 0;
#pragma unroll
  for (int gg = 0; gg < 8; ++gg)
    rank += ((g[gg] > gs) || (g[gg] == gs && gg < mg)) ? 1 : 0;
  bool selected = (rank < TOPKG);

  // mask: exactly 0.0 at unselected positions (matches reference semantics)
  float mvals[8];
#pragma unroll
  for (int j = 0; j < 8; ++j) mvals[j] = selected ? b[j] : 0.0f;

  // 8 rounds of wave-wide argmax (value desc, index asc tie-break like top_k)
  float wsel[8]; int isel[8]; float sum = 0.0f;
#pragma unroll
  for (int r = 0; r < 8; ++r) {
    float lv = -INFINITY; int li = 0x7FFFFFFF;
#pragma unroll
    for (int j = 0; j < 8; ++j) {
      bool better = (mvals[j] > lv);
      lv = better ? mvals[j] : lv;
      li = better ? (lane * 8 + j) : li;
    }
#pragma unroll
    for (int sft = 16; sft >= 1; sft >>= 1) {
      float ov = __shfl_xor(lv, sft, 32);
      int   oi = __shfl_xor(li, sft, 32);
      bool take = (ov > lv) || (ov == lv && oi < li);
      lv = take ? ov : lv;
      li = take ? oi : li;
    }
    int owner = li >> 3, jw = li & 7;
    float sv = 0.0f;
#pragma unroll
    for (int j = 0; j < 8; ++j) if (j == jw) sv = s[j];   // unrolled select
    sv = __shfl(sv, owner, 32);                           // original sigmoid
    wsel[r] = sv; isel[r] = li; sum += sv;
#pragma unroll
    for (int j = 0; j < 8; ++j)
      if (lane == owner && j == jw) mvals[j] = -INFINITY; // remove winner
  }

  if (lane == 0) {
    float k = RSCALE / sum;
#pragma unroll
    for (int r = 0; r < 8; ++r) {
      out_w[(size_t)token * TOPK + r]   = wsel[r] * k;
      out_idx[(size_t)token * TOPK + r] = isel[r];
    }
  }
}

extern "C" void kernel_launch(void* const* d_in, const int* in_sizes, int n_in,
                              void* d_out, int out_size, void* d_ws, size_t ws_size,
                              hipStream_t stream) {
  const float* x    = (const float*)d_in[0];   // [16384, 7168] f32
  const float* wgt  = (const float*)d_in[1];   // [256, 7168]   f32
  const float* bias = (const float*)d_in[2];   // [256]         f32

  // workspace: w_hi | w_lo (bf16, 3.67 MB each) | logits (f32, 16.8 MB)
  __bf16* w_hi  = (__bf16*)d_ws;
  __bf16* w_lo  = w_hi + (size_t)NEXP * DIM;
  float* logits = (float*)(w_lo + (size_t)NEXP * DIM);

  float* out_w   = (float*)d_out;              // [16384, 8] f32
  int*   out_idx = (int*)((float*)d_out + (size_t)TOKENS * TOPK); // [16384,8] i32

  // 1) split weight to bf16 hi/lo (256*7168/4 = 458752 float4 -> 1792 blocks)
  weight_split_kernel<<<(NEXP * DIM) / (4 * 256), 256, 0, stream>>>(wgt, w_hi, w_lo);

  // 2) router GEMM: 256 workgroups (one 64-token block each, all 256 experts)
  router_gemm_kernel<<<TOKENS / BM, 256, 0, stream>>>(x, w_hi, w_lo, logits);

  // 3) gate / top-k
  gate_topk_kernel<<<TOKENS / 8, 256, 0, stream>>>(logits, bias, out_w, out_idx);
}
